// Part_Graph_51539607552376
// MI455X (gfx1250) — matrix operand order repaired and checked by reading, not compiled, not run on hardware
//
#include <hip/hip_runtime.h>
#include <hip/hip_bf16.h>

// ---------- types ----------
typedef __attribute__((ext_vector_type(16))) __bf16        v16bf;
typedef __attribute__((ext_vector_type(8)))  float         v8f;
typedef __attribute__((ext_vector_type(4)))  unsigned int  v4u;

#define HW      9216        // 96*96
#define TILE    128         // pixels per workgroup
#define NTILES  72          // 9216/128
#define NB      8           // batch
#define NN      6           // part nodes
#define OUT_HALF (NN*NB*10*HW)   // 4,423,680 floats per output tensor

// 1/sqrt(1+1e-5) — folded BN scale
#define ISQ 0.99999500003749975f

// ---------- LDS layout (ushort units for bf16 region) ----------
#define A1_OFF   0                         // [128][328] bf16 cat
#define A1_STRIDE 328
#define W1_OFF   (A1_OFF + 128*328)        // [32][320] bf16 nw1
#define H1_OFF   (W1_OFF + 32*320)         // [128][32] bf16 h
#define W2_OFF   (H1_OFF + 128*32)         // [16][32]  bf16 nw2 (rows 10..15 zero)
#define U_OFF    (W2_OFF + 16*32)          // [128][32] bf16 u (cols 20..31 zero)
#define WU1_OFF  (U_OFF + 128*32)          // [32][32]  bf16 uw1 (cols 20..31 zero)
#define H2_OFF   (WU1_OFF + 32*32)         // [128][32] bf16 h2
#define WU2_OFF  (H2_OFF + 128*32)         // [16][32]  bf16 uw2 (rows 10..15 zero)
#define US_TOTAL (WU2_OFF + 16*32)         // = 66,560 ushorts = 133,120 B
// float region (after bf16 region)
#define XHP_OFF  0                         // [10][128] f32
#define MSG_OFF  (XHP_OFF + 10*128)        // [10][128] f32
#define XPN_OFF  (MSG_OFF + 10*128)        // [10][128] f32
#define F_TOTAL  (XPN_OFF + 10*128)
#define SMEM_BYTES (US_TOTAL*2 + F_TOTAL*4)   // 148,480 B

// ---------- helpers ----------
__device__ __forceinline__ unsigned short f2bf(float f) {
    unsigned int u = __float_as_uint(f);
    u += 0x7FFFu + ((u >> 16) & 1u);       // round-to-nearest-even
    return (unsigned short)(u >> 16);
}

// A fragment (16x32 bf16, M across lanes, K packed): per ISA 7.12.2
__device__ __forceinline__ v16bf load_frag_a(const unsigned short* base, int row0,
                                             int kb, int stride, int lane) {
    const int hi = lane >> 4;
    const unsigned short* p = base + (row0 + (lane & 15)) * stride + kb + hi * 8;
    union { v16bf v; v4u q[2]; } u;
    u.q[0] = *(const v4u*)(p);        // K = kb + hi*8 + 0..7
    u.q[1] = *(const v4u*)(p + 16);   // K = kb + 16 + hi*8 + 0..7
    return u.v;
}

// B fragment (32x16 bf16, N across lanes, K packed). Weights stored [N][K] row-major.
__device__ __forceinline__ v16bf load_frag_b(const unsigned short* base, int n0,
                                             int kb, int stride, int lane) {
    const unsigned short* p = base + (n0 + (lane & 15)) * stride + kb + (lane >> 4) * 8;
    union { v16bf v; v4u q[2]; } u;
    u.q[0] = *(const v4u*)(p);        // K = kb + hi*8 + 0..7
    u.q[1] = *(const v4u*)(p + 16);   // K = kb + 16 + hi*8 + 0..7
    return u.v;
}

__device__ __forceinline__ v8f wmma_bf16(v16bf a, v16bf b, v8f c) {
    return __builtin_amdgcn_wmma_f32_16x16x32_bf16(false, a, false, b, (short)0, c,
                                                   false, false);
}

union Acc { v8f v; float f[8]; };

// ---------- kernel ----------
__global__ __launch_bounds__(256)
void part_graph_kernel(const float* __restrict__ p_fea, const float* __restrict__ xp,
                       const float* __restrict__ xhp,
                       const float* __restrict__ nw1, const float* __restrict__ ng1,
                       const float* __restrict__ nb1,
                       const float* __restrict__ nw2, const float* __restrict__ ng2,
                       const float* __restrict__ nb2,
                       const float* __restrict__ uw1, const float* __restrict__ ug1,
                       const float* __restrict__ ub1,
                       const float* __restrict__ uw2, const float* __restrict__ ug2,
                       const float* __restrict__ ub2,
                       float* __restrict__ out) {
    const int tile = blockIdx.x;   // 0..71
    const int b    = blockIdx.y;   // 0..7
    const int n    = blockIdx.z;   // 0..5
    const int tid  = threadIdx.x;
    const int lane = tid & 31;
    const int wave = tid >> 5;     // 0..7 -> M-tile
    const int base = tile * TILE;  // pixel base within image

    extern __shared__ __align__(16) unsigned char smem_raw[];
    unsigned short* S  = (unsigned short*)smem_raw;
    unsigned short* A1 = S + A1_OFF;
    unsigned short* W1 = S + W1_OFF;
    unsigned short* H1 = S + H1_OFF;
    unsigned short* W2 = S + W2_OFF;
    unsigned short* U  = S + U_OFF;
    unsigned short* WU1= S + WU1_OFF;
    unsigned short* H2 = S + H2_OFF;
    unsigned short* WU2= S + WU2_OFF;
    float* F   = (float*)(smem_raw + US_TOTAL * 2);
    float* XHP = F + XHP_OFF;
    float* MSG = F + MSG_OFF;
    float* XPN = F + XPN_OFF;

    // ---- stage weights (zero-padded) ----
    for (int i = tid; i < 32 * 320; i += 256) W1[i] = 0;
    for (int i = tid; i < 16 * 32;  i += 256) W2[i] = 0;
    for (int i = tid; i < 32 * 32;  i += 256) WU1[i] = 0;
    for (int i = tid; i < 16 * 32;  i += 256) WU2[i] = 0;
    __syncthreads();
    for (int i = tid; i < 32 * 306; i += 256) {
        int o = i / 306, c = i % 306;
        W1[o * 320 + c] = f2bf(nw1[(size_t)n * 32 * 306 + i]);
    }
    for (int i = tid; i < 10 * 32; i += 256) W2[i]  = f2bf(nw2[(size_t)n * 320 + i]);
    for (int i = tid; i < 32 * 20; i += 256) {
        int o = i / 20, c = i % 20;
        WU1[o * 32 + c] = f2bf(uw1[(size_t)n * 640 + i]);
    }
    for (int i = tid; i < 10 * 32; i += 256) WU2[i] = f2bf(uw2[(size_t)n * 320 + i]);

    // ---- stage A1 = leave-one-out concat, bf16, pixel-major ----
    // zero K-pad columns 306..319
    for (int i = tid; i < 128 * 14; i += 256) {
        int p = i / 14, c = 306 + i % 14;
        A1[p * A1_STRIDE + c] = 0;
    }
    {
        const int pix  = tid & 127;
        const int half = tid >> 7;                 // two channels per iteration
        const float* pf = p_fea + (size_t)b * 256 * HW + base;
        for (int c0 = 0; c0 < 306; c0 += 2) {
            int c = c0 + half;
            if (c < 306) {
                float v;
                if (c < 256) {
                    v = pf[(size_t)c * HW + pix];
                } else {
                    int k = c - 256, idx = k / 10, sub = k % 10;
                    int i = idx + (idx >= n ? 1 : 0);   // skip node n
                    v = xp[(((size_t)i * NB + b) * 10 + sub) * HW + base + pix];
                }
                A1[pix * A1_STRIDE + c] = f2bf(v);
            }
        }
    }

    // ---- stage xp -> U cols 0..9 (bf16), xhp -> XHP (f32); zero U cols 20..31 ----
    for (int i = tid; i < 10 * 128; i += 256) {
        int o = i >> 7, p = i & 127;
        size_t g = (((size_t)n * NB + b) * 10 + o) * HW + base + p;
        U[p * 32 + o] = f2bf(xp[g]);
        XHP[o * 128 + p] = xhp[g];
    }
    for (int i = tid; i < 128 * 12; i += 256) {
        int p = i / 12, c = 20 + i % 12;
        U[p * 32 + c] = 0;
    }
    __syncthreads();

    const int nlo = lane & 15;
    const int hiL = lane >> 4;
    const int m0  = wave * 16;

    // ================= GEMM1: 306(->320) -> 32 =================
    Acc acc0, acc1; acc0.v = {}; acc1.v = {};
    #pragma unroll
    for (int kb = 0; kb < 320; kb += 32) {
        v16bf a  = load_frag_a(A1, m0, kb, A1_STRIDE, lane);
        v16bf b0 = load_frag_b(W1, 0,  kb, 320, lane);
        v16bf b1 = load_frag_b(W1, 16, kb, 320, lane);
        acc0.v = wmma_bf16(a, b0, acc0.v);
        acc1.v = wmma_bf16(a, b1, acc1.v);
    }
    {   // BN+ReLU -> H1 (bf16)
        float s0 = ng1[n * 32 + nlo] * ISQ,      c0 = nb1[n * 32 + nlo];
        float s1 = ng1[n * 32 + 16 + nlo] * ISQ, c1 = nb1[n * 32 + 16 + nlo];
        #pragma unroll
        for (int r = 0; r < 8; ++r) {
            int m = m0 + hiL * 8 + r;
            H1[m * 32 + nlo]      = f2bf(fmaxf(acc0.f[r] * s0 + c0, 0.f));
            H1[m * 32 + 16 + nlo] = f2bf(fmaxf(acc1.f[r] * s1 + c1, 0.f));
        }
    }
    __syncthreads();

    // ================= GEMM2: 32 -> 10 ; message = xpp + xhp =================
    {
        Acc acc; acc.v = {};
        v16bf a  = load_frag_a(H1, m0, 0, 32, lane);
        v16bf bw = load_frag_b(W2, 0, 0, 32, lane);
        acc.v = wmma_bf16(a, bw, acc.v);
        if (nlo < 10) {
            float s = ng2[n * 10 + nlo] * ISQ, c = nb2[n * 10 + nlo];
            #pragma unroll
            for (int r = 0; r < 8; ++r) {
                int m = m0 + hiL * 8 + r;
                float xpp = fmaxf(acc.f[r] * s + c, 0.f);
                float msg = xpp + XHP[nlo * 128 + m];
                MSG[nlo * 128 + m] = msg;            // f32 for output
                U[m * 32 + 10 + nlo] = f2bf(msg);    // bf16 for GEMM3
            }
        }
    }
    __syncthreads();

    // ================= GEMM3: 20(->32) -> 32 =================
    Acc h20, h21; h20.v = {}; h21.v = {};
    {
        v16bf a  = load_frag_a(U, m0, 0, 32, lane);
        v16bf b0 = load_frag_b(WU1, 0,  0, 32, lane);
        v16bf b1 = load_frag_b(WU1, 16, 0, 32, lane);
        h20.v = wmma_bf16(a, b0, h20.v);
        h21.v = wmma_bf16(a, b1, h21.v);
        float s0 = ug1[n * 32 + nlo] * ISQ,      c0 = ub1[n * 32 + nlo];
        float s1 = ug1[n * 32 + 16 + nlo] * ISQ, c1 = ub1[n * 32 + 16 + nlo];
        #pragma unroll
        for (int r = 0; r < 8; ++r) {
            int m = m0 + hiL * 8 + r;
            H2[m * 32 + nlo]      = f2bf(fmaxf(h20.f[r] * s0 + c0, 0.f));
            H2[m * 32 + 16 + nlo] = f2bf(fmaxf(h21.f[r] * s1 + c1, 0.f));
        }
    }
    __syncthreads();

    // ================= GEMM4: 32 -> 10 =================
    {
        Acc acc; acc.v = {};
        v16bf a  = load_frag_a(H2, m0, 0, 32, lane);
        v16bf bw = load_frag_b(WU2, 0, 0, 32, lane);
        acc.v = wmma_bf16(a, bw, acc.v);
        if (nlo < 10) {
            float s = ug2[n * 10 + nlo] * ISQ, c = ub2[n * 10 + nlo];
            #pragma unroll
            for (int r = 0; r < 8; ++r) {
                int m = m0 + hiL * 8 + r;
                XPN[nlo * 128 + m] = fmaxf(acc.f[r] * s + c, 0.f);
            }
        }
    }
    __syncthreads();

    // ---- coalesced output: xp_new then message ----
    const size_t obase = (((size_t)n * NB + b) * 10) * HW + base;
    for (int i = tid; i < 10 * 128; i += 256) {
        int o = i >> 7, p = i & 127;
        out[obase + (size_t)o * HW + p]            = XPN[i];
        out[OUT_HALF + obase + (size_t)o * HW + p] = MSG[i];
    }
}

// ---------- host ----------
extern "C" void kernel_launch(void* const* d_in, const int* in_sizes, int n_in,
                              void* d_out, int out_size, void* d_ws, size_t ws_size,
                              hipStream_t stream) {
    (void)in_sizes; (void)n_in; (void)out_size; (void)d_ws; (void)ws_size;
    const float* p_fea = (const float*)d_in[0];
    const float* xp    = (const float*)d_in[1];
    const float* xhp   = (const float*)d_in[2];
    const float* nw1   = (const float*)d_in[3];
    const float* ng1   = (const float*)d_in[4];
    const float* nb1   = (const float*)d_in[5];
    const float* nw2   = (const float*)d_in[6];
    const float* ng2   = (const float*)d_in[7];
    const float* nb2   = (const float*)d_in[8];
    const float* uw1   = (const float*)d_in[9];
    const float* ug1   = (const float*)d_in[10];
    const float* ub1   = (const float*)d_in[11];
    const float* uw2   = (const float*)d_in[12];
    const float* ug2   = (const float*)d_in[13];
    const float* ub2   = (const float*)d_in[14];
    float* out = (float*)d_out;

    static_assert(SMEM_BYTES <= 320 * 1024, "LDS over budget");
    (void)hipFuncSetAttribute((const void*)part_graph_kernel,
                              hipFuncAttributeMaxDynamicSharedMemorySize, SMEM_BYTES);

    dim3 grid(NTILES, NB, NN);   // 72 x 8 x 6 = 3456 workgroups
    dim3 block(256);             // 8 wave32
    part_graph_kernel<<<grid, block, SMEM_BYTES, stream>>>(
        p_fea, xp, xhp, nw1, ng1, nb1, nw2, ng2, nb2,
        uw1, ug1, ub1, uw2, ug2, ub2, out);
}
